// DeepBiLSTM_75471165325774
// MI455X (gfx1250) — compile-verified, matrix operand order checked by reading, not run on hardware
//
#include <hip/hip_runtime.h>
#include <hip/hip_bf16.h>
#include <math.h>

// Problem constants
#define BB 64
#define TT 128
#define DD 1024
#define D2 2048   // 2*D (GEMM K)
#define D4 4096   // 4*D (GEMM N, 4 gates)

typedef __attribute__((ext_vector_type(16))) __bf16 v16bf;
typedef __attribute__((ext_vector_type(8)))  float  v8f;
typedef unsigned int u32x4 __attribute__((ext_vector_type(4)));

union ABFrag { v16bf v; u32x4 q[2]; };

// ---------------------------------------------------------------------------
// Convert fp32 W [2048][4096] (k-major) -> bf16 Wt [4096][2048] (n-major).
// Coalesced reads over n; scattered bf16 writes are absorbed by L2 (Wt stays
// resident in the 192MB L2 for all 256 step-GEMMs).
// ---------------------------------------------------------------------------
__global__ void convert_w_kernel(const float* __restrict__ W,
                                 __bf16* __restrict__ Wt) {
    int idx = blockIdx.x * 256 + threadIdx.x;   // 0 .. 2048*4096-1
    int n = idx & (D4 - 1);
    int k = idx >> 12;
    Wt[(size_t)n * D2 + k] = (__bf16)W[(size_t)k * D4 + n];
}

// Zero the recurrent state region (xh1,xh2,c1,c2 contiguous) and the output
// accumulator. Runs every call => deterministic graph replay.
__global__ void zero_init_kernel(u32x4* __restrict__ s, int n4s,
                                 u32x4* __restrict__ o, int n4o) {
    int i = blockIdx.x * 256 + threadIdx.x;
    u32x4 z = {0u, 0u, 0u, 0u};
    if (i < n4s) s[i] = z;
    if (i < n4o) o[i] = z;
}

// Per-step embedding gather: xh1[b][0:D] = bf16(emb[inputs[b][t]])
__global__ void gather_embed_kernel(const int* __restrict__ inputs,
                                    const float* __restrict__ emb,
                                    __bf16* __restrict__ xh1, int t) {
    int idx = blockIdx.x * 256 + threadIdx.x;   // 0..65535
    int b = idx >> 10;
    int d = idx & (DD - 1);
    int tok = inputs[b * TT + t];
    xh1[b * D2 + d] = (__bf16)emb[(size_t)tok * DD + d];
}

// ---------------------------------------------------------------------------
// One LSTM layer for one timestep, fused GEMM + cell update.
//   gates[64][4096] = xh[64][2048](bf16) @ Wt^T + bias   (bf16 WMMA, f32 acc)
// Grid: 64 WGs, WG owns hidden slice d0 = blockIdx.x*16 (gate cols
// {q*1024 + d0 .. +15} for q=0..3). 8 waves: wave w -> gate q=w&3,
// M rows [(w>>2)*32, +32) as two 16x16 tiles sharing one B fragment.
//
// A/B tiles are DOUBLE-BUFFERED in LDS and filled with
// GLOBAL_LOAD_ASYNC_TO_LDS_B128 (ASYNCcnt, no VGPR round-trip): iteration i
// issues loads for buffer (i+1)&1 and waits asynccnt<=2 so only the
// one-iteration-old loads must have landed -> memory latency overlaps WMMA.
// ---------------------------------------------------------------------------
__global__ void __launch_bounds__(256)
lstm_step_kernel(__bf16* __restrict__ xh,          // [64][2048]: [x | h]
                 const __bf16* __restrict__ Wt,    // [4096][2048] n-major
                 const float* __restrict__ bias,   // [4096]
                 float* __restrict__ c,            // [64][1024]
                 __bf16* __restrict__ h_fwd,       // layer1: xh2 (unmasked h1n); else null
                 float* __restrict__ out,          // layer2: d_out accum; else null
                 const int* __restrict__ lengths,
                 int t) {
    // Two tile buffers (each: A 64x32 bf16 = 4KB, then B 64x32 bf16 = 4KB),
    // overlaid with the 16KB gate-exchange buffer used after the K loop.
    __shared__ __align__(16) unsigned char smem[16384];
    __bf16* sT = (__bf16*)smem;            // tiles: buf q at elems q*4096
    float*  sG = (float*)smem;             // gates 4 x 64 x 16 f32 (reused)

    const int d0   = blockIdx.x * 16;
    const int tid  = threadIdx.x;
    const int wave = tid >> 5;             // wave32
    const int lane = tid & 31;
    const int g    = wave & 3;             // gate quadrant
    const int mb   = (wave >> 2) * 32;     // row-group base
    const int lcol = lane & 15;
    const int lhalf = lane >> 4;

    // Staging assignment: each thread moves 8 bf16 (b128) of A and of B
    // per K-iteration, directly memory->LDS via the async path.
    const int ldRow = tid >> 2;            // 0..63
    const int ldK   = (tid & 3) * 8;       // 0,8,16,24
    const size_t aSrcBase = (size_t)ldRow * D2 + ldK;
    const int nGlobal = (ldRow >> 4) * DD + d0 + (ldRow & 15);
    const size_t bSrcBase = (size_t)nGlobal * D2 + ldK;

    // LDS destination byte offsets per buffer (generic LDS pointer low 32
    // bits == LDS offset per ISA aperture layout).
    unsigned ldsA[2], ldsB[2];
    ldsA[0] = (unsigned)(uintptr_t)(sT + 0 * 4096 + ldRow * 32 + ldK);
    ldsB[0] = (unsigned)(uintptr_t)(sT + 0 * 4096 + 2048 + ldRow * 32 + ldK);
    ldsA[1] = (unsigned)(uintptr_t)(sT + 1 * 4096 + ldRow * 32 + ldK);
    ldsB[1] = (unsigned)(uintptr_t)(sT + 1 * 4096 + 2048 + ldRow * 32 + ldK);

    v8f acc0 = {0.f, 0.f, 0.f, 0.f, 0.f, 0.f, 0.f, 0.f};
    v8f acc1 = {0.f, 0.f, 0.f, 0.f, 0.f, 0.f, 0.f, 0.f};

    // Prologue: kick off tile 0 into buffer 0.
    {
        unsigned long long gA = (unsigned long long)(uintptr_t)(xh + aSrcBase);
        unsigned long long gB = (unsigned long long)(uintptr_t)(Wt + bSrcBase);
        asm volatile("global_load_async_to_lds_b128 %0, %1, off"
                     :: "v"(ldsA[0]), "v"(gA) : "memory");
        asm volatile("global_load_async_to_lds_b128 %0, %1, off"
                     :: "v"(ldsB[0]), "v"(gB) : "memory");
    }

    for (int k0 = 0; k0 < D2; k0 += 32) {
        const int cur = (k0 >> 5) & 1;
        if (k0 + 32 < D2) {
            // Issue next tile into the other buffer (last read 2 barriers ago),
            // then wait only for the CURRENT buffer's (older) loads.
            unsigned long long gA =
                (unsigned long long)(uintptr_t)(xh + aSrcBase + k0 + 32);
            unsigned long long gB =
                (unsigned long long)(uintptr_t)(Wt + bSrcBase + k0 + 32);
            asm volatile("global_load_async_to_lds_b128 %0, %1, off"
                         :: "v"(ldsA[cur ^ 1]), "v"(gA) : "memory");
            asm volatile("global_load_async_to_lds_b128 %0, %1, off"
                         :: "v"(ldsB[cur ^ 1]), "v"(gB) : "memory");
            __builtin_prefetch(Wt + bSrcBase + k0 + 64, 0, 1);  // B tile k0+64
            asm volatile("s_wait_asynccnt 0x2" ::: "memory");
        } else {
            asm volatile("s_wait_asynccnt 0x0" ::: "memory");
        }
        __syncthreads();   // current buffer visible to all waves

        const __bf16* base = sT + cur * 4096;
        ABFrag af0, af1, bf;
        const __bf16* bp = base + 2048 + (g * 16 + lcol) * 32 + lhalf * 8;
        bf.q[0] = *(const u32x4*)(bp);
        bf.q[1] = *(const u32x4*)(bp + 16);
        const __bf16* ap0 = base + (mb + lcol) * 32 + lhalf * 8;
        af0.q[0] = *(const u32x4*)(ap0);
        af0.q[1] = *(const u32x4*)(ap0 + 16);
        const __bf16* ap1 = ap0 + 16 * 32;
        af1.q[0] = *(const u32x4*)(ap1);
        af1.q[1] = *(const u32x4*)(ap1 + 16);

        acc0 = __builtin_amdgcn_wmma_f32_16x16x32_bf16(
            false, af0.v, false, bf.v, (short)0, acc0, false, false);
        acc1 = __builtin_amdgcn_wmma_f32_16x16x32_bf16(
            false, af1.v, false, bf.v, (short)0, acc1, false, false);
        __syncthreads();   // all reads of this buffer done before it is refilled
    }

    // Scatter accumulators to LDS gate buffer: sG[gate][row][col]
#pragma unroll
    for (int r = 0; r < 8; ++r) {
        int row0 = mb + lhalf * 8 + r;
        sG[(g * 64 + row0) * 16 + lcol] = acc0[r];
        int row1 = mb + 16 + lhalf * 8 + r;
        sG[(g * 64 + row1) * 16 + lcol] = acc1[r];
    }
    __syncthreads();

    // Cell update: 64 rows x 16 cols, 4 elements per thread.
    for (int e = tid; e < 1024; e += 256) {
        int b   = e >> 4;
        int col = e & 15;
        int d   = d0 + col;
        float gi = sG[(0 * 64 + b) * 16 + col] + bias[0 * DD + d];
        float gj = sG[(1 * 64 + b) * 16 + col] + bias[1 * DD + d];
        float gf = sG[(2 * 64 + b) * 16 + col] + bias[2 * DD + d];
        float go = sG[(3 * 64 + b) * 16 + col] + bias[3 * DD + d];
        float si = 1.f / (1.f + __expf(-gi));
        float sf = 1.f / (1.f + __expf(-gf));
        float so = 1.f / (1.f + __expf(-go));
        float tj = tanhf(gj);
        float cold = c[b * DD + d];
        float cn   = cold * sf + si * tj;
        float hn   = tanhf(cn) * so;
        float m    = (t < lengths[b]) ? 1.f : 0.f;
        float hold = (float)xh[b * D2 + DD + d];
        c[b * DD + d]        = m * cn + (1.f - m) * cold;
        xh[b * D2 + DD + d]  = (__bf16)(m * hn + (1.f - m) * hold);  // recurrent h
        if (h_fwd) h_fwd[b * D2 + d] = (__bf16)hn;                   // layer2 input (unmasked)
        if (out)   out[b * DD + d]  += m * hn;                       // sum_t h2n * m
    }
}

// ---------------------------------------------------------------------------
// Workspace layout (bytes):
//   [0,        16MB)  Wt1 bf16  [4096][2048]
//   [16MB,     32MB)  Wt2 bf16
//   [32MB  +0      )  xh1 bf16  [64][2048]   (256KB)
//          +256KB     xh2 bf16  [64][2048]   (256KB)
//          +512KB     c1  f32   [64][1024]   (256KB)
//          +768KB     c2  f32   [64][1024]   (256KB)
// ---------------------------------------------------------------------------
extern "C" void kernel_launch(void* const* d_in, const int* in_sizes, int n_in,
                              void* d_out, int out_size, void* d_ws, size_t ws_size,
                              hipStream_t stream) {
    const int*   inputs  = (const int*)d_in[0];
    const int*   lengths = (const int*)d_in[1];
    const float* emb     = (const float*)d_in[2];
    const float* W1      = (const float*)d_in[3];
    const float* b1      = (const float*)d_in[4];
    const float* W2      = (const float*)d_in[5];
    const float* b2      = (const float*)d_in[6];
    float* out = (float*)d_out;

    char* ws = (char*)d_ws;
    const size_t WT_BYTES = (size_t)D4 * D2 * 2;      // 16 MB
    __bf16* Wt1 = (__bf16*)(ws);
    __bf16* Wt2 = (__bf16*)(ws + WT_BYTES);
    char* state = ws + 2 * WT_BYTES;
    __bf16* xh1 = (__bf16*)(state);
    __bf16* xh2 = (__bf16*)(state + 256 * 1024);
    float*  c1  = (float*)(state + 512 * 1024);
    float*  c2  = (float*)(state + 768 * 1024);

    // One-time (per call) weight convert+transpose to bf16.
    convert_w_kernel<<<(D2 * D4) / 256, 256, 0, stream>>>(W1, Wt1);
    convert_w_kernel<<<(D2 * D4) / 256, 256, 0, stream>>>(W2, Wt2);

    // Zero state region (1 MB) and output accumulator (256 KB).
    zero_init_kernel<<<256, 256, 0, stream>>>((u32x4*)state, (1024 * 1024) / 16,
                                              (u32x4*)out, (BB * DD * 4) / 16);

    for (int t = 0; t < TT; ++t) {
        gather_embed_kernel<<<(BB * DD) / 256, 256, 0, stream>>>(inputs, emb, xh1, t);
        // Layer 1: consumes [x_t | h1], produces h1n -> xh2 front half.
        lstm_step_kernel<<<DD / 16, 256, 0, stream>>>(xh1, Wt1, b1, c1,
                                                      xh2, nullptr, lengths, t);
        // Layer 2: consumes [h1n | h2], accumulates masked h2n into out.
        lstm_step_kernel<<<DD / 16, 256, 0, stream>>>(xh2, Wt2, b2, c2,
                                                      nullptr, out, lengths, t);
    }
}
